// GNN_MoRec_18494129176905
// MI455X (gfx1250) — compile-verified, matrix operand order checked by reading, not run on hardware
//
#include <hip/hip_runtime.h>
#include <math.h>

#define DH   128      // D == H == 128 (fixed by reference)
#define NREL 8
#define FRAG_ELEMS 16                        // bf16 elems per lane per fragment
#define FRAGS_PER_MAT (4 * 8)                // 4 k-steps x 8 n-tiles
#define MAT_ELEMS (FRAGS_PER_MAT * 32 * FRAG_ELEMS)   // 16384 bf16 = one 128x128 matrix

typedef __attribute__((ext_vector_type(16))) __bf16 v16bf;
typedef __attribute__((ext_vector_type(8)))  float  v8f;

// ISA 7.12.2: 16-bit A/B fragment K index for element e, lane-half base
// (lanes 0-15 -> K base 0, lanes 16-31 -> K base 8).
__device__ __forceinline__ int kidx(int e, int kbase) {
  return kbase + e + ((e >= 8) ? 8 : 0);
}

// ---------- monotonic float<->uint encoding for atomic segment-max ----------
__device__ __forceinline__ unsigned fenc(float f) {
  unsigned u = __float_as_uint(f);
  return (u & 0x80000000u) ? ~u : (u | 0x80000000u);
}
__device__ __forceinline__ float fdec(unsigned u) {
  u = (u & 0x80000000u) ? (u & 0x7FFFFFFFu) : ~u;
  return __uint_as_float(u);
}

// ---------- weight pre-swizzle: f32 [128][128] -> bf16 in WMMA B-fragment order ----
// out layout: frag f = kk*8+nt; elem index = (f*32 + lane)*16 + e
__global__ void k_swizzle(const float* __restrict__ in, __bf16* __restrict__ out) {
  int i = blockIdx.x * blockDim.x + threadIdx.x;   // [0, 16384)
  if (i >= MAT_ELEMS) return;
  int e    = i & 15;
  int lane = (i >> 4) & 31;
  int nt   = (i >> 9) & 7;
  int kk   = i >> 12;
  int kbase = (lane < 16) ? 0 : 8;
  int k = kk * 32 + kidx(e, kbase);
  int n = nt * 16 + (lane & 15);
  out[i] = (__bf16)in[k * DH + n];
}

// ---------- embedding gather: x[n, :] = emb[nodeTypes[n], :] ----------
__global__ void k_gather(const int* __restrict__ nt, const float* __restrict__ emb,
                         float* __restrict__ x, int N) {
  int i = blockIdx.x * blockDim.x + threadIdx.x;
  if (i >= N * DH) return;
  int node = i >> 7, f = i & (DH - 1);
  x[i] = emb[nt[node] * DH + f];
}

// ---------- per-relation pre-aggregation: agg[dst] += x[src], cnt[dst]++ ----------
__global__ void k_scatter(const int* __restrict__ ei, const int* __restrict__ ea,
                          const float* __restrict__ x, float* __restrict__ agg,
                          float* __restrict__ cnt, int E, int rel) {
  int e    = blockIdx.x * (blockDim.x >> 5) + (threadIdx.x >> 5);
  int lane = threadIdx.x & 31;
  if (e >= E) return;
  if (ea[e] != rel) return;              // uniform across the wave
  int src = ei[e];
  int dst = ei[E + e];
  const float4 v = *(const float4*)(x + (size_t)src * DH + lane * 4);
  float* a = agg + (size_t)dst * DH + lane * 4;
  atomicAdd(a + 0, v.x);
  atomicAdd(a + 1, v.y);
  atomicAdd(a + 2, v.z);
  atomicAdd(a + 3, v.w);
  if (lane == 0) atomicAdd(cnt + dst, 1.0f);
}

// ---------- WMMA GEMM: C[N,128] (+)= rowscale(A)[N,128] @ B[128,128] ----------
// Bsw: pre-swizzled bf16 weight (fragment order).
// MODE 0: C = A@B + bias;  MODE 1: C += (A * 1/max(cnt,1)) @ B.
// block = 512 threads (16 wave32); block tile = 128 rows x 128 cols;
// wave tile = 16 rows x 64 cols (wave w: rows (w>>1)*16.., cols (w&1)*64..).
// A-global and B-LDS fragment loads are double-buffered across k-steps.
template <int MODE>
__global__ void k_gemm(const float* __restrict__ A, const __bf16* __restrict__ Bsw,
                       float* __restrict__ C, const float* __restrict__ bias,
                       const float* __restrict__ cnt, int N) {
  __shared__ __align__(32) __bf16 Bl[MAT_ELEMS];   // 32 KB

  // straight 32KB memcpy: global_load_b128 -> ds_store_b128
  {
    const int4* src = (const int4*)Bsw;
    int4* dst = (int4*)Bl;
    const int n16 = MAT_ELEMS * (int)sizeof(__bf16) / 16;   // 2048
    for (int i = threadIdx.x; i < n16; i += blockDim.x) dst[i] = src[i];
  }
  __syncthreads();

  int wave  = threadIdx.x >> 5;
  int lane  = threadIdx.x & 31;
  int half  = lane & 15;
  int kbase = (lane < 16) ? 0 : 8;
  int wrow  = wave >> 1;                   // 0..7
  int nb    = (wave & 1) * 4;              // n-tile base: 0 or 4
  int m0    = blockIdx.x * 128 + wrow * 16;

  int  mrow  = m0 + half;                  // both lane halves carry rows m0..m0+15
  bool valid = (mrow < N);
  int  mc    = valid ? mrow : 0;
  float scale = valid ? 1.0f : 0.0f;
  if (MODE == 1 && valid) scale = 1.0f / fmaxf(cnt[mc], 1.0f);  // fused mean-normalize
  const float* arow = A + (size_t)mc * DH + kbase;

  const __bf16* blbase = Bl + (size_t)lane * FRAG_ELEMS;
#define BFRAG(kk, nt) (*(const v16bf*)(blbase + (size_t)((kk) * 8 + (nt)) * 32 * FRAG_ELEMS))

  float4 a[2][4];
  v16bf  bf[2][4];

  // prologue: stage k-step 0
#pragma unroll
  for (int q = 0; q < 4; ++q) a[0][q] = *(const float4*)(arow + ((q >> 1) * 16 + (q & 1) * 4));
#pragma unroll
  for (int nt = 0; nt < 4; ++nt) bf[0][nt] = BFRAG(0, nb + nt);

  v8f acc[4] = {};
#pragma unroll
  for (int kk = 0; kk < 4; ++kk) {                 // K = 128 in steps of 32
    const int cur = kk & 1, nxt = cur ^ 1;
    if (kk < 3) {
      // issue next k-step's loads before this step's WMMA burst
      const float* apn = arow + (kk + 1) * 32;
#pragma unroll
      for (int q = 0; q < 4; ++q) a[nxt][q] = *(const float4*)(apn + ((q >> 1) * 16 + (q & 1) * 4));
#pragma unroll
      for (int nt = 0; nt < 4; ++nt) bf[nxt][nt] = BFRAG(kk + 1, nb + nt);
    }

    v16bf af;
#pragma unroll
    for (int q = 0; q < 4; ++q) {
      af[q * 4 + 0] = (__bf16)(a[cur][q].x * scale);
      af[q * 4 + 1] = (__bf16)(a[cur][q].y * scale);
      af[q * 4 + 2] = (__bf16)(a[cur][q].z * scale);
      af[q * 4 + 3] = (__bf16)(a[cur][q].w * scale);
    }

#pragma unroll
    for (int nt = 0; nt < 4; ++nt)
      acc[nt] = __builtin_amdgcn_wmma_f32_16x16x32_bf16(
          false, af, false, bf[cur][nt], (short)0, acc[nt], false, false);
  }
#undef BFRAG

  // C/D layout: VGPR v holds row m0 + v (+8 for lanes 16-31), col = nt*16 + (lane&15)
  int rbase = m0 + ((lane < 16) ? 0 : 8);
  if (m0 + 128 <= N) {
    // full tile: branch-free stores
#pragma unroll
    for (int nt = 0; nt < 4; ++nt) {
      int col = (nb + nt) * 16 + half;
#pragma unroll
      for (int v = 0; v < 8; ++v) {
        size_t off = (size_t)(rbase + v) * DH + col;
        if (MODE == 0) C[off] = acc[nt][v] + bias[col];
        else           C[off] += acc[nt][v];
      }
    }
  } else {
#pragma unroll
    for (int nt = 0; nt < 4; ++nt) {
      int col = (nb + nt) * 16 + half;
#pragma unroll
      for (int v = 0; v < 8; ++v) {
        int row = rbase + v;
        if (row < N) {
          size_t off = (size_t)row * DH + col;
          if (MODE == 0) C[off] = acc[nt][v] + bias[col];
          else           C[off] += acc[nt][v];
        }
      }
    }
  }
}

// ---------- elementwise ReLU ----------
__global__ void k_relu(float* __restrict__ h, int n) {
  int i = blockIdx.x * blockDim.x + threadIdx.x;
  if (i < n) h[i] = fmaxf(h[i], 0.0f);
}

// ---------- attention logits + segment max (wave per node) ----------
__global__ void k_logits(const float* __restrict__ h, const float* __restrict__ aw,
                         const int* __restrict__ bs, float* __restrict__ logit,
                         unsigned* __restrict__ menc, int N) {
  int n    = blockIdx.x * (blockDim.x >> 5) + (threadIdx.x >> 5);
  int lane = threadIdx.x & 31;
  if (n >= N) return;
  const float4 hv = *(const float4*)(h + (size_t)n * DH + lane * 4);
  const float4 wv = *(const float4*)(aw + lane * 4);
  float s = hv.x * wv.x + hv.y * wv.y + hv.z * wv.z + hv.w * wv.w;
  for (int off = 16; off; off >>= 1) s += __shfl_down(s, off, 32);
  if (lane == 0) {
    logit[n] = s;
    atomicMax(menc + bs[n], fenc(s));
  }
}

// ---------- e = exp(logit - segmax), z[g] += e ----------
__global__ void k_expz(float* __restrict__ logit, const int* __restrict__ bs,
                       const unsigned* __restrict__ menc, float* __restrict__ z, int N) {
  int n = blockIdx.x * blockDim.x + threadIdx.x;
  if (n >= N) return;
  int g = bs[n];
  float e = __expf(logit[n] - fdec(menc[g]));
  logit[n] = e;                       // reuse as e-buffer
  atomicAdd(z + g, e);
}

// ---------- gacc[g] += (e/z[g]) * h[n]  (wave per node) ----------
__global__ void k_accum(const float* __restrict__ h, const float* __restrict__ ew,
                        const int* __restrict__ bs, const float* __restrict__ z,
                        float* __restrict__ gacc, int N) {
  int n    = blockIdx.x * (blockDim.x >> 5) + (threadIdx.x >> 5);
  int lane = threadIdx.x & 31;
  if (n >= N) return;
  int g = bs[n];
  float att = ew[n] / z[g];
  const float4 hv = *(const float4*)(h + (size_t)n * DH + lane * 4);
  float* a = gacc + (size_t)g * DH + lane * 4;
  atomicAdd(a + 0, att * hv.x);
  atomicAdd(a + 1, att * hv.y);
  atomicAdd(a + 2, att * hv.z);
  atomicAdd(a + 3, att * hv.w);
}

// ---------- out[g] = sigmoid(gacc[g] . lin_w + lin_b) (wave per graph) ----------
__global__ void k_final(const float* __restrict__ gacc, const float* __restrict__ lw,
                        const float* __restrict__ lb, float* __restrict__ out, int G) {
  int g = blockIdx.x;
  if (g >= G) return;
  int lane = threadIdx.x;
  float s = 0.0f;
  for (int j = lane; j < DH; j += 32) s += gacc[(size_t)g * DH + j] * lw[j];
  for (int off = 16; off; off >>= 1) s += __shfl_down(s, off, 32);
  if (lane == 0) out[g] = 1.0f / (1.0f + __expf(-(s + lb[0])));
}

// ---------- one RGCN layer ----------
// wswL: 9 pre-swizzled matrices (0 = root, 1..8 = W_r)
static void run_layer(const float* xin, const __bf16* wswL, const float* b,
                      float* out, float* agg, float* cnt,
                      const int* ei, const int* ea, int N, int E, hipStream_t s) {
  dim3 blk(256);
  dim3 gblk(512);
  int gemm_blocks = (N + 127) / 128;
  k_gemm<0><<<gemm_blocks, gblk, 0, s>>>(xin, wswL, out, b, nullptr, N);
  for (int r = 0; r < NREL; ++r) {
    hipMemsetAsync(agg, 0, (size_t)N * DH * sizeof(float), s);
    hipMemsetAsync(cnt, 0, (size_t)N * sizeof(float), s);
    k_scatter<<<(E + 7) / 8, blk, 0, s>>>(ei, ea, xin, agg, cnt, E, r);
    k_gemm<1><<<gemm_blocks, gblk, 0, s>>>(agg, wswL + (size_t)(1 + r) * MAT_ELEMS,
                                           out, nullptr, cnt, N);
  }
  int tot = N * DH;
  k_relu<<<(tot + 255) / 256, blk, 0, s>>>(out, tot);
}

extern "C" void kernel_launch(void* const* d_in, const int* in_sizes, int n_in,
                              void* d_out, int out_size, void* d_ws, size_t ws_size,
                              hipStream_t stream) {
  const int*   nodeTypes = (const int*)d_in[0];
  const int*   edge_idx  = (const int*)d_in[1];
  const int*   edge_attr = (const int*)d_in[2];
  const int*   bs        = (const int*)d_in[3];
  const float* emb       = (const float*)d_in[5];
  const float* W1        = (const float*)d_in[6];
  const float* root1     = (const float*)d_in[7];
  const float* b1        = (const float*)d_in[8];
  const float* W2        = (const float*)d_in[9];
  const float* root2     = (const float*)d_in[10];
  const float* b2        = (const float*)d_in[11];
  const float* att_w     = (const float*)d_in[12];
  const float* lin_w     = (const float*)d_in[13];
  const float* lin_b     = (const float*)d_in[14];
  float* out = (float*)d_out;

  const int N = in_sizes[0];
  const int E = in_sizes[2];
  const int G = out_size;            // output is [G,1]

  // workspace: x (reused as h2) | h1 | agg | cnt | logits | menc | z | gacc | wsw(bf16)
  float*    x     = (float*)d_ws;
  float*    h1    = x + (size_t)N * DH;
  float*    agg   = h1 + (size_t)N * DH;
  float*    cnt   = agg + (size_t)N * DH;
  float*    logit = cnt + N;
  unsigned* menc  = (unsigned*)(logit + N);
  float*    z     = (float*)(menc + G);
  float*    gacc  = z + G;
  __bf16*   wsw1  = (__bf16*)(gacc + (size_t)G * DH);   // 9 matrices, layer 1
  __bf16*   wsw2  = wsw1 + (size_t)9 * MAT_ELEMS;       // 9 matrices, layer 2
  float*    h2    = x;               // x dead after layer 1 -> alias

  dim3 blk(256);
  const int swz_blocks = (MAT_ELEMS + 255) / 256;

  // pre-swizzle all 18 weight matrices into WMMA fragment order (bf16)
  k_swizzle<<<swz_blocks, blk, 0, stream>>>(root1, wsw1);
  for (int r = 0; r < NREL; ++r)
    k_swizzle<<<swz_blocks, blk, 0, stream>>>(W1 + (size_t)r * DH * DH,
                                              wsw1 + (size_t)(1 + r) * MAT_ELEMS);
  k_swizzle<<<swz_blocks, blk, 0, stream>>>(root2, wsw2);
  for (int r = 0; r < NREL; ++r)
    k_swizzle<<<swz_blocks, blk, 0, stream>>>(W2 + (size_t)r * DH * DH,
                                              wsw2 + (size_t)(1 + r) * MAT_ELEMS);

  // x = emb[nodeTypes]
  k_gather<<<((N * DH) + 255) / 256, blk, 0, stream>>>(nodeTypes, emb, x, N);

  // two RGCN layers (WMMA GEMMs + edge scatter)
  run_layer(x,  wsw1, b1, h1, agg, cnt, edge_idx, edge_attr, N, E, stream);
  run_layer(h1, wsw2, b2, h2, agg, cnt, edge_idx, edge_attr, N, E, stream);

  // attention pooling + readout
  hipMemsetAsync(menc, 0, (size_t)G * sizeof(unsigned), stream);
  hipMemsetAsync(z,    0, (size_t)G * sizeof(float), stream);
  hipMemsetAsync(gacc, 0, (size_t)G * DH * sizeof(float), stream);
  k_logits<<<(N + 7) / 8, blk, 0, stream>>>(h2, att_w, bs, logit, menc, N);
  k_expz  <<<(N + 255) / 256, blk, 0, stream>>>(logit, bs, menc, z, N);
  k_accum <<<(N + 7) / 8, blk, 0, stream>>>(h2, logit, bs, z, gacc, N);
  k_final <<<G, dim3(32), 0, stream>>>(gacc, lin_w, lin_b, out, G);
}
